// WeavePredictor_43765716746313
// MI455X (gfx1250) — matrix operand order repaired and checked by reading, not compiled, not run on hardware
//
#include <hip/hip_runtime.h>
#include <hip/hip_bf16.h>

// ---------------------------------------------------------------------------
// WeavePredictor on MI455X (gfx1250, wave32).
//
// Roofline: ~105 GFLOP total, dominated by (E=480000,128)x(128,128) GEMMs.
// Streaming each big fp32 operand once per GEMM => few GB of HBM traffic
// => bandwidth bound at 23.3 TB/s. Matmuls use v_wmma_f32_16x16x32_bf16
// (bf16 in, f32 accumulate); weights are tiny and pre-packed on device into
// per-lane fragment layout so B loads are two global_load_b128 per fragment.
// A tiles are staged fp32->bf16 into LDS and read back with ds_load_b128 in
// contiguous k order (lane half h holds k=16h..16h+15), same as packed B. All 8 B fragments of a k-step
// are loaded into registers before the 8 WMMAs so load latency overlaps
// matrix issue (single clause + decreasing s_wait_loadcnt).
// ---------------------------------------------------------------------------

#define HF 128            // hidden / feature width (all GEMMs have N=128)
#define NTILE 8           // 128 / 16 output tiles per wave
#define BLK_ROWS 64       // M rows per workgroup (4 waves x 16)
#define GEMM_THREADS 128  // 4 wave32 waves
#define LDS_PITCH 40      // ushorts per staged row: 32 data + 8 pad (80B)

typedef __attribute__((ext_vector_type(16))) __bf16 v16bf;
typedef __attribute__((ext_vector_type(8)))  float  v8f;

__device__ __forceinline__ unsigned short f2bf(float x) {
  union { float f; unsigned u; } c; c.f = x;
  unsigned r = c.u + 0x7FFFu + ((c.u >> 16) & 1u);   // round-to-nearest-even
  return (unsigned short)(r >> 16);
}

union Frag16 { v16bf v; uint4 q[2]; };

// ---------------------------------------------------------------------------
// Pack a (K x 128) row-major fp32 weight into bf16 WMMA B-fragments:
// layout [kblk][ntile][lane][16 halves]; lane 0-15: n=lane, k=kblk*32+h;
// lane 16-31: n=lane&15, k=kblk*32+16+h  (per-lane 32B contiguous).
// ---------------------------------------------------------------------------
__global__ void pack_weight(const float* __restrict__ W,
                            unsigned short* __restrict__ Bp, int K) {
  int idx = blockIdx.x * blockDim.x + threadIdx.x;
  if (idx >= K * HF) return;
  int h    = idx & 15;
  int lane = (idx >> 4) & 31;
  int nt   = (idx >> 9) & 7;
  int kblk = idx >> 12;
  int n = nt * 16 + (lane & 15);
  int k = kblk * 32 + ((lane & 16) ? 16 : 0) + h;
  Bp[idx] = f2bf(W[k * HF + n]);
}

// ---------------------------------------------------------------------------
// C(M,128) = act( A(M,K) @ Wpacked + bias [+ C_old] )
// act: 0 = none, 1 = relu, 2 = tanh.  accumulate: read-add previous C.
// ---------------------------------------------------------------------------
__global__ __launch_bounds__(GEMM_THREADS)
void gemm128_wmma(const float* __restrict__ A, long lda,
                  const unsigned short* __restrict__ Bp,
                  const float* __restrict__ bias,
                  float* __restrict__ C, long ldc,
                  int M, int K, int act, int accumulate) {
  __shared__ unsigned short smA[BLK_ROWS * LDS_PITCH];
  const int tid  = threadIdx.x;
  const int lane = tid & 31;
  const int wave = tid >> 5;
  const int row0 = blockIdx.x * BLK_ROWS;

  v8f acc[NTILE];
#pragma unroll
  for (int t = 0; t < NTILE; ++t) acc[t] = (v8f){0.f,0.f,0.f,0.f,0.f,0.f,0.f,0.f};

  const int halfsel = (lane & 16) ? 16 : 0;         // A lane k-offset
  const int mloc    = wave * 16 + (lane & 15);      // staged row for A frag

  for (int k0 = 0; k0 < K; k0 += 32) {
    // --- stage 64x32 fp32 -> bf16 into LDS (512 float4 slots / 128 thr) ---
#pragma unroll
    for (int i = 0; i < 4; ++i) {
      int lin = tid + i * GEMM_THREADS;
      int r   = lin >> 3;          // 8 float4 per 32-col row
      int kc  = (lin & 7) * 4;
      float4 f = make_float4(0.f, 0.f, 0.f, 0.f);
      int gr = row0 + r;
      if (gr < M) f = *(const float4*)(A + (long)gr * lda + k0 + kc);
      ushort4 h;
      h.x = f2bf(f.x); h.y = f2bf(f.y); h.z = f2bf(f.z); h.w = f2bf(f.w);
      *(ushort4*)(&smA[r * LDS_PITCH + kc]) = h;
    }
    __syncthreads();

    // --- A fragment: two ds_load_b128, contiguous k = 16*half + s (as packed B) ---
    Frag16 a;
    a.q[0] = *(const uint4*)(&smA[mloc * LDS_PITCH + halfsel]);
    a.q[1] = *(const uint4*)(&smA[mloc * LDS_PITCH + 8 + halfsel]);

    // --- all 8 B fragments first (one load clause), then 8 WMMAs ---
    const unsigned short* bp =
        Bp + (long)(k0 >> 5) * (NTILE * 32 * 16) + (long)lane * 16;
    Frag16 b[NTILE];
#pragma unroll
    for (int nt = 0; nt < NTILE; ++nt) {
      b[nt].q[0] = *(const uint4*)(bp + nt * 512);
      b[nt].q[1] = *(const uint4*)(bp + nt * 512 + 8);
    }
#pragma unroll
    for (int nt = 0; nt < NTILE; ++nt) {
      acc[nt] = __builtin_amdgcn_wmma_f32_16x16x32_bf16(
          false, a.v, false, b[nt].v, (short)0, acc[nt], false, false);
    }
    __syncthreads();
  }

  // --- epilogue: C VGPR j holds M=j (lanes 0-15) / M=8+j (lanes 16-31) ---
  const int n0    = lane & 15;
  const int mbase = row0 + wave * 16 + ((lane & 16) ? 8 : 0);
#pragma unroll
  for (int nt = 0; nt < NTILE; ++nt) {
    int n = nt * 16 + n0;
    float bv = bias ? bias[n] : 0.0f;
#pragma unroll
    for (int j = 0; j < 8; ++j) {
      int m = mbase + j;
      if (m < M) {
        float v = acc[nt][j] + bv;
        if (accumulate) v += C[(long)m * ldc + n];
        if (act == 1)      v = fmaxf(v, 0.0f);
        else if (act == 2) v = tanhf(v);
        C[(long)m * ldc + n] = v;
      }
    }
  }
}

// ---------------------------------------------------------------------------
// Elementwise / scatter / reduction helpers
// ---------------------------------------------------------------------------
__global__ void zero_f32(float* p, long n) {
  long i = (long)blockIdx.x * blockDim.x + threadIdx.x;
  if (i < n) p[i] = 0.0f;
}

__global__ void scatter_edges(const float* __restrict__ e2n,
                              const int* __restrict__ dst,
                              float* __restrict__ agg, int E) {
  long i = (long)blockIdx.x * blockDim.x + threadIdx.x;
  if (i >= (long)E * HF) return;
  int e = (int)(i >> 7), f = (int)(i & 127);
  atomicAdd(&agg[(long)dst[e] * HF + f], e2n[i]);
}

__global__ void edge_pair(const float* __restrict__ left,
                          const float* __restrict__ right,
                          const int* __restrict__ src,
                          const int* __restrict__ dst,
                          float* __restrict__ first,
                          float* __restrict__ second, int E) {
  long i = (long)blockIdx.x * blockDim.x + threadIdx.x;
  if (i >= (long)E * HF) return;
  int e = (int)(i >> 7), f = (int)(i & 127);
  int s = src[e], d = dst[e];
  float ls = left[(long)s * HF + f], rd = right[(long)d * HF + f];
  float rs = right[(long)s * HF + f], ld = left[(long)d * HF + f];
  first[i]  = fmaxf(ls + rd, 0.0f);
  second[i] = fmaxf(rs + ld, 0.0f);
}

__global__ void bn_partial(const float* __restrict__ h, float* __restrict__ ps,
                           float* __restrict__ ps2, int V, int rowsPerBlk) {
  int f = threadIdx.x, b = blockIdx.x;
  float s = 0.f, s2 = 0.f;
  for (int r = 0; r < rowsPerBlk; ++r) {
    int v = b * rowsPerBlk + r;
    if (v < V) { float x = h[(long)v * HF + f]; s += x; s2 += x * x; }
  }
  ps[b * HF + f] = s; ps2[b * HF + f] = s2;
}

__global__ void bn_final(const float* __restrict__ ps, const float* __restrict__ ps2,
                         const float* __restrict__ gamma, const float* __restrict__ beta,
                         float* __restrict__ scale, float* __restrict__ shift,
                         int nblk, int V) {
  int f = threadIdx.x;
  float s = 0.f, s2 = 0.f;
  for (int b = 0; b < nblk; ++b) { s += ps[b * HF + f]; s2 += ps2[b * HF + f]; }
  float mu = s / (float)V;
  float var = s2 / (float)V - mu * mu;
  float sc = gamma[f] * rsqrtf(var + 1e-5f);
  scale[f] = sc;
  shift[f] = beta[f] - mu * sc;
}

__device__ const float kMEANS[11] = {-1.645f,-1.08f,-0.739f,-0.468f,-0.228f,0.0f,
                                     0.228f,0.468f,0.739f,1.08f,1.645f};
__device__ const float kSTDS[11]  = {0.283f,0.17f,0.134f,0.118f,0.114f,0.114f,
                                     0.114f,0.118f,0.134f,0.17f,0.283f};

__global__ void hist_scatter(const float* __restrict__ h,
                             const float* __restrict__ scale,
                             const float* __restrict__ shift,
                             const int* __restrict__ gid,
                             float* __restrict__ gfeats, int V) {
  long i = (long)blockIdx.x * blockDim.x + threadIdx.x;
  if (i >= (long)V * HF) return;
  int v = (int)(i >> 7), f = (int)(i & 127);
  float x = h[i] * scale[f] + shift[f];
  float w[11]; float s = 0.f;
#pragma unroll
  for (int k = 0; k < 11; ++k) {
    float t = (x - kMEANS[k]) / kSTDS[k];
    w[k] = __expf(-0.5f * t * t);
    s += w[k];
  }
  float inv = 1.0f / s;
  float* base = &gfeats[(long)gid[v] * (HF * 11) + (long)f * 11];
#pragma unroll
  for (int k = 0; k < 11; ++k) atomicAdd(base + k, w[k] * inv);
}

__global__ void pred_kernel(const float* __restrict__ gout,
                            const float* __restrict__ pw,
                            const float* __restrict__ pb,
                            float* __restrict__ out, int G) {
  int g = blockIdx.x * blockDim.x + threadIdx.x;
  if (g >= G) return;
  float s = pb[0];
  for (int h = 0; h < HF; ++h) s += gout[(long)g * HF + h] * pw[h];
  out[g] = s;
}

// ---------------------------------------------------------------------------
// Host-side orchestration (all launches on `stream`, graph-capture safe).
// ---------------------------------------------------------------------------
static inline dim3 g1(long n, int t) { return dim3((unsigned)((n + t - 1) / t)); }

extern "C" void kernel_launch(void* const* d_in, const int* in_sizes, int n_in,
                              void* d_out, int out_size, void* d_ws, size_t ws_size,
                              hipStream_t stream) {
  (void)in_sizes; (void)n_in; (void)out_size; (void)ws_size;
  const int V = 30000, E = 480000, G = 256;

  // Input order: setup_inputs() insertion order, params flattened as a pytree
  // in insertion order: layers[0..1]{n2n,e2n,upd_n,left,right,e2e,upd_e}(w,b),
  // then n2g(w,b), bn_gamma, bn_beta, to_out(w,b), pred(w,b).
  const float* node = (const float*)d_in[0];
  const float* edge = (const float*)d_in[1];
  const int*   src  = (const int*)d_in[2];
  const int*   dst  = (const int*)d_in[3];
  const int*   gid  = (const int*)d_in[4];
  auto LP = [&](int layer, int off) { return (const float*)d_in[5 + layer * 14 + off]; };
  // per-layer offsets: 0 n2n.w 1 n2n.b 2 e2n.w 3 e2n.b 4 updn.w 5 updn.b
  //                    6 left.w 7 left.b 8 right.w 9 right.b 10 e2e.w 11 e2e.b
  //                    12 upde.w 13 upde.b
  const float* n2g_w  = (const float*)d_in[33];
  const float* n2g_b  = (const float*)d_in[34];
  const float* bn_g   = (const float*)d_in[35];
  const float* bn_b   = (const float*)d_in[36];
  const float* to_w   = (const float*)d_in[37];
  const float* to_b   = (const float*)d_in[38];
  const float* pred_w = (const float*)d_in[39];
  const float* pred_b = (const float*)d_in[40];
  float* out = (float*)d_out;

  // --- workspace carve-out ---
  char* ws = (char*)d_ws;
  size_t off = 0;
  auto alloc = [&](size_t bytes) -> void* {
    off = (off + 255) & ~(size_t)255;
    void* p = (void*)(ws + off);
    off += bytes;
    return p;
  };
  const size_t PK = (size_t)HF * HF * sizeof(unsigned short);   // 128x128 slice
  unsigned short* p_l0_n2n   = (unsigned short*)alloc(PK);
  unsigned short* p_l0_e2n   = (unsigned short*)alloc(PK);
  unsigned short* p_l0_updn0 = (unsigned short*)alloc(PK);
  unsigned short* p_l0_updn1 = (unsigned short*)alloc(PK);
  unsigned short* p_l0_left  = (unsigned short*)alloc(PK);
  unsigned short* p_l0_right = (unsigned short*)alloc(PK);
  unsigned short* p_l0_e2e   = (unsigned short*)alloc(PK);
  unsigned short* p_l0_upde0 = (unsigned short*)alloc(PK);
  unsigned short* p_l0_upde1 = (unsigned short*)alloc(PK);
  unsigned short* p_l0_upde2 = (unsigned short*)alloc(PK);
  unsigned short* p_l1_n2n   = (unsigned short*)alloc(PK);
  unsigned short* p_l1_e2n   = (unsigned short*)alloc(PK);
  unsigned short* p_l1_updn0 = (unsigned short*)alloc(PK);
  unsigned short* p_l1_updn1 = (unsigned short*)alloc(PK);
  unsigned short* p_n2g      = (unsigned short*)alloc(PK);
  unsigned short* p_toout    = (unsigned short*)alloc((size_t)1408 * HF * sizeof(unsigned short));

  const size_t NB = (size_t)V * HF * sizeof(float);
  const size_t EB = (size_t)E * HF * sizeof(float);
  float* nn_h   = (float*)alloc(NB);
  float* agg    = (float*)alloc(NB);
  float* node1  = (float*)alloc(NB);
  float* leftB  = (float*)alloc(NB);
  float* rightB = (float*)alloc(NB);
  float* node2  = (float*)alloc(NB);
  float* hB     = (float*)alloc(NB);
  float* e2nB   = (float*)alloc(EB);   // reused: e2n(l0) -> third -> e2n(l1)
  float* firstB = (float*)alloc(EB);
  float* secondB= (float*)alloc(EB);
  float* edge1  = (float*)alloc(EB);
  const int NBN = 256;
  float* psum   = (float*)alloc((size_t)NBN * HF * sizeof(float));
  float* psum2  = (float*)alloc((size_t)NBN * HF * sizeof(float));
  float* scaleB = (float*)alloc(HF * sizeof(float));
  float* shiftB = (float*)alloc(HF * sizeof(float));
  float* gfeats = (float*)alloc((size_t)G * HF * 11 * sizeof(float));
  float* gout   = (float*)alloc((size_t)G * HF * sizeof(float));

  auto pack = [&](const float* W, unsigned short* Bp, int K) {
    pack_weight<<<g1((long)K * HF, 256), 256, 0, stream>>>(W, Bp, K);
  };
  auto gemm = [&](const float* A, long lda, const unsigned short* Bp,
                  const float* bias, float* C, long ldc, int M, int K,
                  int act, int accum) {
    gemm128_wmma<<<dim3((M + BLK_ROWS - 1) / BLK_ROWS), GEMM_THREADS, 0, stream>>>(
        A, lda, Bp, bias, C, ldc, M, K, act, accum);
  };

  // --- pack all weights to bf16 fragment layout ---
  pack(LP(0,0),  p_l0_n2n,   128);
  pack(LP(0,2),  p_l0_e2n,   128);
  pack(LP(0,4),            p_l0_updn0, 128);
  pack(LP(0,4) + 128*HF,   p_l0_updn1, 128);
  pack(LP(0,6),  p_l0_left,  128);
  pack(LP(0,8),  p_l0_right, 128);
  pack(LP(0,10), p_l0_e2e,   128);
  pack(LP(0,12),            p_l0_upde0, 128);
  pack(LP(0,12) + 128*HF,   p_l0_upde1, 128);
  pack(LP(0,12) + 256*HF,   p_l0_upde2, 128);
  pack(LP(1,0),  p_l1_n2n,   128);
  pack(LP(1,2),  p_l1_e2n,   128);
  pack(LP(1,4),            p_l1_updn0, 128);
  pack(LP(1,4) + 128*HF,   p_l1_updn1, 128);
  pack(n2g_w,    p_n2g,      128);
  pack(to_w,     p_toout,    1408);

  const long EN = (long)E * HF, VN = (long)V * HF;

  // ---------------- layer 0 ----------------
  gemm(node, HF, p_l0_n2n, LP(0,1), nn_h, HF, V, 128, /*relu*/1, 0);
  gemm(edge, HF, p_l0_e2n, LP(0,3), e2nB, HF, E, 128, 1, 0);
  zero_f32<<<g1(VN, 256), 256, 0, stream>>>(agg, VN);
  scatter_edges<<<g1(EN, 256), 256, 0, stream>>>(e2nB, dst, agg, E);
  gemm(nn_h, HF, p_l0_updn0, nullptr, node1, HF, V, 128, 0, 0);
  gemm(agg,  HF, p_l0_updn1, LP(0,5), node1, HF, V, 128, 1, 1);
  gemm(node, HF, p_l0_left,  LP(0,7), leftB,  HF, V, 128, 0, 0);
  gemm(node, HF, p_l0_right, LP(0,9), rightB, HF, V, 128, 0, 0);
  edge_pair<<<g1(EN, 256), 256, 0, stream>>>(leftB, rightB, src, dst, firstB, secondB, E);
  gemm(edge, HF, p_l0_e2e, LP(0,11), e2nB, HF, E, 128, 1, 0);   // third (reuses e2nB)
  gemm(firstB,  HF, p_l0_upde0, nullptr,  edge1, HF, E, 128, 0, 0);
  gemm(secondB, HF, p_l0_upde1, nullptr,  edge1, HF, E, 128, 0, 1);
  gemm(e2nB,    HF, p_l0_upde2, LP(0,13), edge1, HF, E, 128, 1, 1);

  // ---------------- layer 1 (node only) ----------------
  gemm(node1, HF, p_l1_n2n, LP(1,1), leftB, HF, V, 128, 1, 0);  // nn_h1 -> leftB
  gemm(edge1, HF, p_l1_e2n, LP(1,3), e2nB,  HF, E, 128, 1, 0);  // e2n1 -> e2nB
  zero_f32<<<g1(VN, 256), 256, 0, stream>>>(agg, VN);
  scatter_edges<<<g1(EN, 256), 256, 0, stream>>>(e2nB, dst, agg, E);
  gemm(leftB, HF, p_l1_updn0, nullptr, node2, HF, V, 128, 0, 0);
  gemm(agg,   HF, p_l1_updn1, LP(1,5), node2, HF, V, 128, 1, 1);

  // ---------------- readout ----------------
  gemm(node2, HF, p_n2g, n2g_b, hB, HF, V, 128, /*tanh*/2, 0);
  int rpb = (V + NBN - 1) / NBN;
  bn_partial<<<dim3(NBN), dim3(HF), 0, stream>>>(hB, psum, psum2, V, rpb);
  bn_final<<<dim3(1), dim3(HF), 0, stream>>>(psum, psum2, bn_g, bn_b, scaleB, shiftB, NBN, V);
  long GN = (long)G * HF * 11;
  zero_f32<<<g1(GN, 256), 256, 0, stream>>>(gfeats, GN);
  hist_scatter<<<g1(VN, 256), 256, 0, stream>>>(hB, scaleB, shiftB, gid, gfeats, V);
  gemm(gfeats, HF * 11, p_toout, to_b, gout, HF, G, 1408, 2, 0);
  pred_kernel<<<dim3(1), dim3(G), 0, stream>>>(gout, pred_w, pred_b, out, G);
}
